// RaMDE_7593502179376
// MI455X (gfx1250) — matrix operand order repaired and checked by reading, not compiled, not run on hardware
//
#include <hip/hip_runtime.h>
#include <hip/hip_bf16.h>

typedef __attribute__((ext_vector_type(16))) _Float16 v16h;
typedef __attribute__((ext_vector_type(8)))  float    v8f;

// Problem constants
constexpr int B_   = 4;
constexpr int N_   = 256;
constexpr int CH_  = 128;
constexpr int CIN_ = 1024;
constexpr int H_   = 120;
constexpr int W_   = 160;
constexpr int SRC_ = 256;
constexpr int MINB = 5;
constexpr int HW_  = H_ * W_;

constexpr size_t FEAT = (size_t)B_ * CH_ * H_ * W_;      // 9,830,400 floats
constexpr size_t DEPN = 2 * FEAT;
constexpr size_t ZPAD = (size_t)16 * HW_;                // zero pad region (floats)

constexpr int RS = 24;     // padded LDS row stride in halfs (16 data + 8 pad)

union V16 {
    v16h v;
    uint4 q[2];
};
union H16 {
    _Float16 h[16];
    uint4 q[2];
};

#if __has_builtin(__builtin_amdgcn_sched_barrier)
#define SCHED_FENCE() __builtin_amdgcn_sched_barrier(0)
#else
#define SCHED_FENCE() asm volatile("" ::: "memory")
#endif

// ---------------------------------------------------------------------------
// rel_embedding: two 1x1 convs (channel matmuls). One block per (b, n).
// ---------------------------------------------------------------------------
__global__ void relmap_kernel(const float* __restrict__ tf,
                              const float* __restrict__ w1, const float* __restrict__ b1,
                              const float* __restrict__ w2, const float* __restrict__ b2,
                              float* __restrict__ rel_map)
{
    int bn = blockIdx.x;
    int b = bn / N_, n = bn % N_;
    int c = threadIdx.x;                    // 128 threads
    __shared__ float colS[CIN_];
    __shared__ float mid[CH_];
    for (int k = c; k < CIN_; k += CH_)
        colS[k] = tf[((size_t)b * CIN_ + k) * N_ + n];
    __syncthreads();
    float s = b1[c];
    const float* wr = w1 + (size_t)c * CIN_;
    for (int k = 0; k < CIN_; ++k) s += wr[k] * colS[k];
    mid[c] = fmaxf(s, 0.0f);
    __syncthreads();
    float o = b2[c];
    const float* w2r = w2 + (size_t)c * CH_;
    for (int k = 0; k < CH_; ++k) o += w2r[k] * mid[k];
    rel_map[((size_t)b * CH_ + c) * SRC_ + n] = o;
}

// ---------------------------------------------------------------------------
// zero fill
// ---------------------------------------------------------------------------
__global__ void zero_kernel(float* __restrict__ p, size_t n)
{
    size_t i = (size_t)blockIdx.x * blockDim.x + threadIdx.x;
    size_t stride = (size_t)gridDim.x * blockDim.x;
    for (; i < n; i += stride) p[i] = 0.0f;
}

// ---------------------------------------------------------------------------
// interp + scatter-add of box patches. One block per (b, n, contrib).
// ---------------------------------------------------------------------------
__global__ void scatter_kernel(const int* __restrict__ bbox,
                               const float* __restrict__ rel_map,
                               float* __restrict__ rel)
{
    int bid = blockIdx.x;
    int contrib = bid & 1;
    int bn = bid >> 1;
    int b = bn / N_, n = bn % N_;
    const int* bx = bbox + ((size_t)b * N_ + n) * 8;
    int sx1 = bx[0] >> 1, sy1 = bx[1] >> 1, sx2 = bx[2] >> 1, sy2 = bx[3] >> 1;
    int ox1 = bx[4] >> 1, oy1 = bx[5] >> 1, ox2 = bx[6] >> 1, oy2 = bx[7] >> 1;
    int sh = sy2 - sy1, sw = sx2 - sx1;
    int oh = oy2 - oy1, ow = ox2 - ox1;
    bool valid = (sh >= MINB) && (sw >= MINB) && (oh >= MINB) && (ow >= MINB);
    if (!valid) return;
    int x1, y1, x2, y2, h;
    if (contrib == 0) { x1 = sx1; y1 = sy1; x2 = sx2; y2 = sy2; h = sh; }
    else              { x1 = ox1; y1 = oy1; x2 = ox2; y2 = oy2; h = oh; }
    if (h < 1) h = 1;
    int c = threadIdx.x;                        // 128 threads = channels
    int ylo = y1 < 0 ? 0 : y1;
    int yhi = y2 > H_ ? H_ : y2;
    int xlo = x1 < 0 ? 0 : x1;
    int xhi = x2 > W_ ? W_ : x2;
    const float* rmrow = rel_map + ((size_t)b * CH_ + c) * SRC_;
    for (int y = ylo; y < yhi; ++y) {
        int idx = (y - y1) * SRC_ / h;
        if (idx > SRC_ - 1) idx = SRC_ - 1;
        float v = rmrow[idx];
        float* dst = rel + (((size_t)b * CH_ + c) * H_ + y) * W_;
        for (int x = xlo; x < xhi; ++x)
            atomicAdd(&dst[x], v);
    }
}

// ---------------------------------------------------------------------------
// dep = concat(img, rel) along channels
// ---------------------------------------------------------------------------
__global__ void concat_kernel(const float* __restrict__ img,
                              const float* __restrict__ rel,
                              float* __restrict__ dep)
{
    size_t i = (size_t)blockIdx.x * blockDim.x + threadIdx.x;
    if (i >= DEPN) return;
    size_t hw = (size_t)H_ * W_;
    size_t per_b = 2 * CH_ * hw;
    int b = (int)(i / per_b);
    size_t r = i - (size_t)b * per_b;
    int cc = (int)(r / hw);
    size_t p = r - (size_t)cc * hw;
    float v = (cc < CH_) ? img[((size_t)b * CH_ + cc) * hw + p]
                         : rel[((size_t)b * CH_ + (cc - CH_)) * hw + p];
    dep[i] = v;
}

// ---------------------------------------------------------------------------
// Pack fp32 OIHW weights into f16 [co][kq=ky*3+kx][ci] (ci placed at coff
// inside a Ctot-wide concatenated channel dim).
// ---------------------------------------------------------------------------
__global__ void pack_w_kernel(const float* __restrict__ w, _Float16* __restrict__ dst,
                              int Cout, int Cin, int Ctot, int coff)
{
    int i = blockIdx.x * 256 + threadIdx.x;
    int total = Cout * Cin * 9;
    if (i >= total) return;
    int kq = i % 9;
    int t  = i / 9;
    int ci = t % Cin;
    int co = t / Cin;
    dst[((size_t)co * 9 + kq) * Ctot + coff + ci] = (_Float16)w[i];
}

// ---------------------------------------------------------------------------
// Implicit-GEMM 3x3 SAME conv with WMMA f32_16x16x32_f16.
//   out[co,p] = relu?( scale*(acc + b0 + b1) + skip )
//   K = (C0 + C1) * 9; in1 optional (channel concat expresses conv sums).
// Block: 320 threads (10 waves). Tile: 64 out-channels x 160 pixels (one row).
// Wave w owns N-subtile w (B fragment loaded once) and all 4 M-subtiles:
// 4 WMMAs per wave per chunk = 40 per block. B staging: one thread per
// (pixel, k-half) unit -> 16 channel-strided immediate-offset global loads
// + 2 ds_store_b128. Double-buffered LDS, one barrier per chunk. A hard
// sched_barrier keeps all fragment loads ahead of the WMMAs so the waitcnt
// pass can use partial dscnt waits (loads overlap matrix execution).
// ---------------------------------------------------------------------------
__global__ __launch_bounds__(320)
void conv3x3_wmma_kernel(const float* __restrict__ in0, int C0,
                         const float* __restrict__ in1, int C1,
                         const _Float16* __restrict__ Wp,
                         const float* __restrict__ bias0,
                         const float* __restrict__ bias1,
                         float scale,
                         const float* __restrict__ skip,
                         int do_relu,
                         const float* __restrict__ zpad,
                         float* __restrict__ out, int Cout)
{
    __shared__ __align__(16) _Float16 As[2][128 * RS];   // 4 msubs * 32 rows
    __shared__ __align__(16) _Float16 Bs[2][320 * RS];   // 10 nsubs * 32 rows

    const int Ctot = C0 + C1;
    const int CC = Ctot >> 5;               // 32-wide channel chunks
    const int nchunks = CC * 9;             // 36 / 72 / 108 : always even

    // pixel tile = one full row
    int pt = blockIdx.x;
    int y  = pt % H_;
    int b  = pt / H_;
    int co0 = blockIdx.y * 64;

    const int tid  = threadIdx.x;           // 0..319
    const int lane = tid & 31;
    const int wv   = tid >> 5;              // 0..9 ; wave wv owns nsub = wv

    // A-stage mapping (threads 0..255): m = tid>>2, 8 consecutive k at (tid&3)*8
    const int am = tid >> 2;
    const int ak = (tid & 3) * 8;
    const int aRow = ((am >> 4) * 32) + (am & 15) + ((ak & 16) ? 16 : 0);
    const int aE   = (ak & 8) ? 8 : 0;
    const bool aAct = tid < 256;
    const _Float16* aSrcBase = Wp + (size_t)(co0 + (am & 63)) * 9 * Ctot + ak;

    // B-stage mapping: one unit per thread: (j = tid>>1, khalf = tid&1)
    const int bj  = tid >> 1;               // 0..159
    const int bkh = tid & 1;                // 0 / 1
    const int bRow = (bj & 15) + ((bj >> 4) << 5) + (bkh ? 16 : 0);

    // per-batch tensor bases (branchless channel-concat select)
    const float* base0 = in0 + (size_t)b * C0 * HW_;
    const float* base1 = (C1 > 0) ? (in1 + (size_t)b * C1 * HW_) : in0;

    v8f acc[4] = {v8f{}, v8f{}, v8f{}, v8f{}};

    // prefetch chunk-walk state
    int pcc = 0, pky = 0, pkx = 0;
    int aOff = 0;                            // chunk * 32 within [kq][ci]

    auto prefetch = [&](uint4& aR, H16& bh) {
        if (aAct) aR = *(const uint4*)(aSrcBase + aOff);
        aOff += 32;
        int cb = (pcc << 5) + (bkh << 4);    // first channel of this 16-run
        bool s0 = cb < C0;
        int cl = s0 ? cb : (cb - C0);
        const float* tb = s0 ? base0 : base1;
        int yy = y + pky - 1;
        int xx = bj + pkx - 1;
        bool ok = (yy >= 0) && (yy < H_) && (xx >= 0) && (xx < W_);
        const float* p0 = ok ? (tb + cl * HW_ + yy * W_ + xx) : zpad;
        float f[16];
#pragma unroll
        for (int i = 0; i < 16; ++i) f[i] = p0[(size_t)i * HW_];
#pragma unroll
        for (int i = 0; i < 16; ++i) bh.h[i] = (_Float16)f[i];
        if (++pcc == CC) {                   // next kq
            pcc = 0;
            if (++pkx == 3) { pkx = 0; ++pky; }
        }
    };

    auto stage = [&](int buf, const uint4& aR, const H16& bh) {
        if (aAct) *(uint4*)&As[buf][aRow * RS + aE] = aR;
        _Float16* d = &Bs[buf][bRow * RS];
        *(uint4*)d = bh.q[0];
        *(uint4*)(d + 8) = bh.q[1];
    };

    auto compute = [&](int buf) {
        const _Float16* Ab = As[buf];
        const _Float16* Bb = Bs[buf];
        V16 bf;
        {
            int br = (wv * 32 + lane) * RS;
            bf.q[0] = *(const uint4*)&Bb[br];
            bf.q[1] = *(const uint4*)&Bb[br + 8];
        }
        V16 a0, a1, a2, a3;
        {
            int ar = lane * RS;
            a0.q[0] = *(const uint4*)&Ab[ar];
            a0.q[1] = *(const uint4*)&Ab[ar + 8];
            a1.q[0] = *(const uint4*)&Ab[ar + 32 * RS];
            a1.q[1] = *(const uint4*)&Ab[ar + 32 * RS + 8];
            a2.q[0] = *(const uint4*)&Ab[ar + 64 * RS];
            a2.q[1] = *(const uint4*)&Ab[ar + 64 * RS + 8];
            a3.q[0] = *(const uint4*)&Ab[ar + 96 * RS];
            a3.q[1] = *(const uint4*)&Ab[ar + 96 * RS + 8];
        }
        // Hard scheduling barrier: nothing may cross, so all five fragments
        // are live here -> distinct registers and partial s_wait_dscnt
        // instead of a full LDS drain before each WMMA.
        SCHED_FENCE();
        acc[0] = __builtin_amdgcn_wmma_f32_16x16x32_f16(
            false, a0.v, false, bf.v, (short)0, acc[0], false, false);
        acc[1] = __builtin_amdgcn_wmma_f32_16x16x32_f16(
            false, a1.v, false, bf.v, (short)0, acc[1], false, false);
        acc[2] = __builtin_amdgcn_wmma_f32_16x16x32_f16(
            false, a2.v, false, bf.v, (short)0, acc[2], false, false);
        acc[3] = __builtin_amdgcn_wmma_f32_16x16x32_f16(
            false, a3.v, false, bf.v, (short)0, acc[3], false, false);
    };

    uint4 aR0, aR1;
    H16 bh0, bh1;

    prefetch(aR0, bh0);                       // chunk 0
    for (int k = 0; k < nchunks; k += 2) {
        prefetch(aR1, bh1);                   // chunk k+1 (n even -> exists)
        stage(0, aR0, bh0);
        __syncthreads();
        compute(0);
        if (k + 2 < nchunks) prefetch(aR0, bh0);   // chunk k+2
        stage(1, aR1, bh1);
        __syncthreads();
        compute(1);
    }

    // ---- epilogue: scale, bias, skip, relu, store ----
    const int nlo = lane & 15;
    const int mhi = (lane >> 4) * 8;
    const int xx = wv * 16 + nlo;             // always < W_
#pragma unroll
    for (int m = 0; m < 4; ++m) {
#pragma unroll
        for (int r = 0; r < 8; ++r) {
            int co = co0 + m * 16 + r + mhi;
            float bsum = (bias0[co] + (bias1 ? bias1[co] : 0.0f)) * scale;
            float v = acc[m][r] * scale + bsum;
            size_t oidx = (((size_t)b * Cout + co) * H_ + y) * W_ + xx;
            if (skip) v += skip[oidx];
            if (do_relu) v = fmaxf(v, 0.0f);
            out[oidx] = v;
        }
    }
}

// ---------------------------------------------------------------------------
// host side
// ---------------------------------------------------------------------------
static inline void* ws_alloc(char* base, size_t& off, size_t bytes)
{
    off = (off + 255) & ~(size_t)255;
    void* p = base + off;
    off += bytes;
    return p;
}

extern "C" void kernel_launch(void* const* d_in, const int* in_sizes, int n_in,
                              void* d_out, int out_size, void* d_ws, size_t ws_size,
                              hipStream_t stream)
{
    (void)in_sizes; (void)n_in; (void)out_size; (void)ws_size;

    const float* ram      = (const float*)d_in[0];   // range_attention_maps (B,CH,H,W)
    const float* tf       = (const float*)d_in[1];   // tuple_features (B,CIN,N)
    const float* rel_w1   = (const float*)d_in[2];
    const float* rel_b1   = (const float*)d_in[3];
    const float* rel_w2   = (const float*)d_in[4];
    const float* rel_b2   = (const float*)d_in[5];
    const float* dmc_w    = (const float*)d_in[6];   // (4,256,128,3,3)
    const float* dmc_b    = (const float*)d_in[7];   // (4,256)
    const float* imc_w01  = (const float*)d_in[8];   // (2,128,256,3,3)
    const float* imc_b01  = (const float*)d_in[9];   // (2,128)
    const float* imc_w23  = (const float*)d_in[10];  // (2,128,128,3,3)
    const float* imc_b23  = (const float*)d_in[11];
    const float* rmc_w01  = (const float*)d_in[12];
    const float* rmc_b01  = (const float*)d_in[13];
    const float* rmc_w23  = (const float*)d_in[14];
    const float* rmc_b23  = (const float*)d_in[15];
    const float* emb_w1   = (const float*)d_in[16];  // (128,256,3,3)
    const float* emb_b1   = (const float*)d_in[17];
    const float* emb_w2   = (const float*)d_in[18];  // (128,128,3,3)
    const float* emb_b2   = (const float*)d_in[19];
    const int*   bbox     = (const int*)d_in[20];    // (B,N,8)

    char* ws = (char*)d_ws;
    size_t off = 0;

    float* zpad    = (float*)ws_alloc(ws, off, ZPAD * sizeof(float));
    float* rel_map = (float*)ws_alloc(ws, off, (size_t)B_ * CH_ * SRC_ * sizeof(float));
    float* rel0    = (float*)ws_alloc(ws, off, FEAT * sizeof(float));
    float* relA    = (float*)ws_alloc(ws, off, FEAT * sizeof(float));
    float* imgA    = (float*)ws_alloc(ws, off, FEAT * sizeof(float));
    float* imgB    = (float*)ws_alloc(ws, off, FEAT * sizeof(float));
    float* dep0    = (float*)ws_alloc(ws, off, DEPN * sizeof(float));
    float* depA    = (float*)ws_alloc(ws, off, DEPN * sizeof(float));

    const size_t DM_SZ  = (size_t)256 * 2304;  // Cout=256, Ctot=256, x9
    const size_t IM_SZ  = (size_t)128 * 3456;  // Cout=128, Ctot=384, x9
    const size_t E1_SZ  = (size_t)128 * 2304;
    const size_t E2_SZ  = (size_t)128 * 1152;
    _Float16* Wp_dm0 = (_Float16*)ws_alloc(ws, off, DM_SZ * sizeof(_Float16));
    _Float16* Wp_dm1 = (_Float16*)ws_alloc(ws, off, DM_SZ * sizeof(_Float16));
    _Float16* Wp_im0 = (_Float16*)ws_alloc(ws, off, IM_SZ * sizeof(_Float16));
    _Float16* Wp_im1 = (_Float16*)ws_alloc(ws, off, IM_SZ * sizeof(_Float16));
    _Float16* Wp_rm0 = (_Float16*)ws_alloc(ws, off, IM_SZ * sizeof(_Float16));
    _Float16* Wp_rm1 = (_Float16*)ws_alloc(ws, off, IM_SZ * sizeof(_Float16));
    _Float16* Wp_e1  = (_Float16*)ws_alloc(ws, off, E1_SZ * sizeof(_Float16));
    _Float16* Wp_e2  = (_Float16*)ws_alloc(ws, off, E2_SZ * sizeof(_Float16));

    // --- rel_embedding ---
    relmap_kernel<<<B_ * N_, CH_, 0, stream>>>(tf, rel_w1, rel_b1, rel_w2, rel_b2, rel_map);

    // --- zero pad region (conv halo) + rel scatter target ---
    zero_kernel<<<1200, 256, 0, stream>>>(zpad, ZPAD);
    zero_kernel<<<2048, 256, 0, stream>>>(rel0, FEAT);
    scatter_kernel<<<B_ * N_ * 2, CH_, 0, stream>>>(bbox, rel_map, rel0);

    // --- dep0 = concat(img0, rel0) ---
    {
        int blocks = (int)((DEPN + 255) / 256);
        concat_kernel<<<blocks, 256, 0, stream>>>(ram, rel0, dep0);
    }

    // --- pack all conv weights to f16 fragment-friendly layout ---
    auto pack = [&](const float* w, _Float16* dst, int Cout, int Cin, int Ctot, int coff) {
        int total = Cout * Cin * 9;
        pack_w_kernel<<<(total + 255) / 256, 256, 0, stream>>>(w, dst, Cout, Cin, Ctot, coff);
    };
    const size_t DMW   = (size_t)256 * 128 * 9;   // one dmc_w slice
    const size_t IMW01 = (size_t)128 * 256 * 9;   // one imc_w01 slice
    const size_t IMW23 = (size_t)128 * 128 * 9;   // one imc_w23 slice
    pack(dmc_w + 0 * DMW, Wp_dm0, 256, 128, 256, 0);
    pack(dmc_w + 2 * DMW, Wp_dm0, 256, 128, 256, 128);
    pack(dmc_w + 1 * DMW, Wp_dm1, 256, 128, 256, 0);
    pack(dmc_w + 3 * DMW, Wp_dm1, 256, 128, 256, 128);
    pack(imc_w01 + 0 * IMW01, Wp_im0, 128, 256, 384, 0);
    pack(imc_w23 + 0 * IMW23, Wp_im0, 128, 128, 384, 256);
    pack(imc_w01 + 1 * IMW01, Wp_im1, 128, 256, 384, 0);
    pack(imc_w23 + 1 * IMW23, Wp_im1, 128, 128, 384, 256);
    pack(rmc_w01 + 0 * IMW01, Wp_rm0, 128, 256, 384, 0);
    pack(rmc_w23 + 0 * IMW23, Wp_rm0, 128, 128, 384, 256);
    pack(rmc_w01 + 1 * IMW01, Wp_rm1, 128, 256, 384, 0);
    pack(rmc_w23 + 1 * IMW23, Wp_rm1, 128, 128, 384, 256);
    pack(emb_w1, Wp_e1, 128, 256, 256, 0);
    pack(emb_w2, Wp_e2, 128, 128, 128, 0);

    // --- TriGraph message passing via WMMA implicit-GEMM convs ---
    dim3 gridP(B_ * H_, 4);   // Cout = 256
    dim3 gridC(B_ * H_, 2);   // Cout = 128

    // iteration 0 (reads img0=ram, rel0, dep0)
    conv3x3_wmma_kernel<<<gridP, 320, 0, stream>>>(
        ram, CH_, rel0, CH_, Wp_dm0, dmc_b + 0, dmc_b + 512, 0.5f, dep0, 1, zpad, depA, 256);
    conv3x3_wmma_kernel<<<gridC, 320, 0, stream>>>(
        dep0, 2 * CH_, rel0, CH_, Wp_im0, imc_b01 + 0, imc_b23 + 0, 0.5f, ram, 1, zpad, imgA, 128);
    conv3x3_wmma_kernel<<<gridC, 320, 0, stream>>>(
        dep0, 2 * CH_, ram, CH_, Wp_rm0, rmc_b01 + 0, rmc_b23 + 0, 0.5f, rel0, 1, zpad, relA, 128);

    // iteration 1 (reads imgA, relA, depA)
    conv3x3_wmma_kernel<<<gridP, 320, 0, stream>>>(
        imgA, CH_, relA, CH_, Wp_dm1, dmc_b + 256, dmc_b + 768, 0.5f, depA, 1, zpad, dep0, 256);
    conv3x3_wmma_kernel<<<gridC, 320, 0, stream>>>(
        depA, 2 * CH_, relA, CH_, Wp_im1, imc_b01 + 128, imc_b23 + 128, 0.5f, imgA, 1, zpad, imgB, 128);
    conv3x3_wmma_kernel<<<gridC, 320, 0, stream>>>(
        depA, 2 * CH_, imgA, CH_, Wp_rm1, rmc_b01 + 128, rmc_b23 + 128, 0.5f, relA, 1, zpad, rel0, 128);

    // --- final embedding: conv(2CH->CH)+ReLU then conv(CH->CH) ---
    // imgA is dead after iteration 1; reuse as the intermediate.
    conv3x3_wmma_kernel<<<gridC, 320, 0, stream>>>(
        dep0, 2 * CH_, nullptr, 0, Wp_e1, emb_b1, nullptr, 1.0f, nullptr, 1, zpad, imgA, 128);
    conv3x3_wmma_kernel<<<gridC, 320, 0, stream>>>(
        imgA, CH_, nullptr, 0, Wp_e2, emb_b2, nullptr, 1.0f, nullptr, 0, zpad, (float*)d_out, 128);
}
